// LSTM_49057116455326
// MI455X (gfx1250) — compile-verified, matrix operand order checked by reading, not run on hardware
//
#include <hip/hip_runtime.h>

typedef unsigned short u16;
typedef unsigned int u32;
typedef unsigned long long u64;
typedef __attribute__((ext_vector_type(16))) __bf16 v16bf;
typedef __attribute__((ext_vector_type(8)))  float  v8f;

#define H    256          // hidden size
#define G4   1024         // 4*H gate width
#define TT   512          // sequence length
#define BT   16           // batch tile per block (WMMA M)
#define IND  5            // input dim
#define NWM  (G4 * H)     // elements per weight matrix

// round-to-nearest-even f32 -> bf16 (as raw u16)
__device__ __forceinline__ u16 f2bf(float f) {
    u32 u = __float_as_uint(f);
    u += 0x7FFFu + ((u >> 16) & 1u);
    return (u16)(u >> 16);
}

union Frag { uint4 q[2]; v16bf v; };

// ---------------------------------------------------------------------------
// Kernel 1: convert the three 1024x256 fp32 weight matrices to bf16 in d_ws.
// Layout [g][k] row-major, identical to the source, so row g of W is the
// K-column of the WMMA B-matrix for gate-output g.
// ---------------------------------------------------------------------------
__global__ __launch_bounds__(256)
void cvt_bf16_kernel(const float* __restrict__ Whh0,
                     const float* __restrict__ Wih1,
                     const float* __restrict__ Whh1,
                     u16* __restrict__ out) {
    int i = blockIdx.x * blockDim.x + threadIdx.x;
    if (i < NWM)          out[i] = f2bf(Whh0[i]);
    else if (i < 2 * NWM) out[i] = f2bf(Wih1[i - NWM]);
    else if (i < 3 * NWM) out[i] = f2bf(Whh1[i - 2 * NWM]);
}

// ---------------------------------------------------------------------------
// Kernel 2: persistent fused 2-layer LSTM scan + output head.
// grid = 16 blocks (one 16-row batch tile each), block = 256 threads (8 waves).
// Weights are re-fetched from global each step (L2-resident, 1.5 MB shared).
// An opaque zero element-offset (re-defined by empty asm every iteration)
// defeats LICM of the loop-invariant weight loads WITHOUT breaking global
// address-space inference, so they lower to global_load_b128 (LOADcnt only),
// not flat_load_b128 (LOADcnt+DScnt).
// ---------------------------------------------------------------------------
__global__ __launch_bounds__(256)
void lstm_persistent_kernel(const float* __restrict__ x,      // [B][T][5]
                            const float* __restrict__ Wih0,   // [1024][5] fp32
                            const float* __restrict__ bih0,
                            const float* __restrict__ bhh0,
                            const float* __restrict__ bih1,
                            const float* __restrict__ bhh1,
                            const float* __restrict__ Wh,     // [3][256]
                            const float* __restrict__ bhb,    // [3]
                            const u16*   __restrict__ Wr0,    // Whh0 bf16 [1024][256]
                            const u16*   __restrict__ Wi1,    // Wih1 bf16
                            const u16*   __restrict__ Wr1,    // Whh1 bf16
                            float* __restrict__ out)          // [B][1][3]
{
    __shared__ u16   h0t[BT * H];        // layer-0 hidden tile, bf16
    __shared__ u16   h1t[BT * H];        // layer-1 hidden tile, bf16
    __shared__ float gatesLDS[BT * G4];  // gate pre-activations (GEMM out)
    __shared__ float hfin[BT * H];       // fp32 h1 at final step
    __shared__ float xls[BT * 8];        // x tile for current t (padded 5->8)

    const int tid   = threadIdx.x;
    const int lane  = tid & 31;
    const int wave  = tid >> 5;
    const int bBase = blockIdx.x * BT;

    // ---- zero LDS hidden state
    for (int i = tid; i < BT * H; i += 256) { h0t[i] = 0; h1t[i] = 0; }

    // ---- per-thread cell constants: this thread owns hidden column j
    const int j = tid;
    float wi0[4][IND], b0c[4], b1c[4];
#pragma unroll
    for (int g = 0; g < 4; ++g) {
#pragma unroll
        for (int d = 0; d < IND; ++d) wi0[g][d] = Wih0[(g * H + j) * IND + d];
        b0c[g] = bih0[g * H + j] + bhh0[g * H + j];
        b1c[g] = bih1[g * H + j] + bhh1[g * H + j];
    }
    float c0[BT], c1[BT];
#pragma unroll
    for (int b = 0; b < BT; ++b) { c0[b] = 0.f; c1[b] = 0.f; }

    // ---- WMMA fragment addressing (ISA 16-bit A 16x32 / f32 C-D layouts)
    const int mA    = lane & 15;            // A: lane -> row M
    const int koffA = (lane >> 4) * 8;      // A/B: per-half K sub-offset
    const int nB    = lane & 15;            // B/D: lane -> column N
    const int mBase = (lane >> 4) * 8;      // D: VGPR r -> row mBase + r
    const int g0b   = wave * 128;           // this wave's 128 gate columns

    const v8f zacc = {};

    // opaque zero offset (elements); refreshed each t to defeat LICM while
    // keeping the kernarg pointer chain (=> global address space) intact
    u64 woff = 0;

    __syncthreads();

    for (int t = 0; t < TT; ++t) {
        asm volatile("" : "+s"(woff));
        const u16* __restrict__ wr0 = Wr0 + woff;
        const u16* __restrict__ wi1 = Wi1 + woff;
        const u16* __restrict__ wr1 = Wr1 + woff;

        // ---- stage x[b, t, :] for this tile
        if (tid < BT * IND) {
            int b = tid / IND, d = tid % IND;
            xls[b * 8 + d] = x[((size_t)(bBase + b) * TT + t) * IND + d];
        }

        // ================= layer 0 recurrent GEMM: h0t @ Whh0^T ==========
        v8f acc[8];
#pragma unroll
        for (int s = 0; s < 8; ++s) acc[s] = zacc;
#pragma unroll
        for (int kk = 0; kk < H / 32; ++kk) {
            const int k0 = kk * 32;
            Frag a;
            a.q[0] = *(const uint4*)&h0t[mA * H + k0 + koffA];
            a.q[1] = *(const uint4*)&h0t[mA * H + k0 + 16 + koffA];
#pragma unroll
            for (int s = 0; s < 8; ++s) {
                const int g = g0b + s * 16 + nB;
                Frag bf;
                bf.q[0] = *(const uint4*)&wr0[g * H + k0 + koffA];
                bf.q[1] = *(const uint4*)&wr0[g * H + k0 + 16 + koffA];
                acc[s] = __builtin_amdgcn_wmma_f32_16x16x32_bf16(
                    false, a.v, false, bf.v, (short)0, acc[s], false, false);
            }
        }
#pragma unroll
        for (int s = 0; s < 8; ++s) {
            const int g = g0b + s * 16 + nB;
#pragma unroll
            for (int r = 0; r < 8; ++r)
                gatesLDS[(mBase + r) * G4 + g] = acc[s][r];
        }
        __syncthreads();

        // ================= layer 0 cell (adds bias + x@Wih0^T) ===========
#pragma unroll
        for (int b = 0; b < BT; ++b) {
            float pg[4];
#pragma unroll
            for (int g = 0; g < 4; ++g) {
                float p = gatesLDS[b * G4 + g * H + j] + b0c[g];
#pragma unroll
                for (int d = 0; d < IND; ++d) p += xls[b * 8 + d] * wi0[g][d];
                pg[g] = p;
            }
            float ig = 1.f / (1.f + __expf(-pg[0]));
            float fg = 1.f / (1.f + __expf(-pg[1]));
            float gg = tanhf(pg[2]);
            float og = 1.f / (1.f + __expf(-pg[3]));
            c0[b] = fg * c0[b] + ig * gg;
            float h = og * tanhf(c0[b]);
            h0t[b * H + j] = f2bf(h);
        }
        __syncthreads();

        // ====== layer 1 GEMMs: h0t @ Wih1^T  +  h1t @ Whh1^T =============
#pragma unroll
        for (int s = 0; s < 8; ++s) acc[s] = zacc;
#pragma unroll
        for (int pass = 0; pass < 2; ++pass) {
            const u16* __restrict__ Wp = pass ? wr1 : wi1;
            const u16* __restrict__ hp = pass ? h1t : h0t;
#pragma unroll
            for (int kk = 0; kk < H / 32; ++kk) {
                const int k0 = kk * 32;
                Frag a;
                a.q[0] = *(const uint4*)&hp[mA * H + k0 + koffA];
                a.q[1] = *(const uint4*)&hp[mA * H + k0 + 16 + koffA];
#pragma unroll
                for (int s = 0; s < 8; ++s) {
                    const int g = g0b + s * 16 + nB;
                    Frag bf;
                    bf.q[0] = *(const uint4*)&Wp[g * H + k0 + koffA];
                    bf.q[1] = *(const uint4*)&Wp[g * H + k0 + 16 + koffA];
                    acc[s] = __builtin_amdgcn_wmma_f32_16x16x32_bf16(
                        false, a.v, false, bf.v, (short)0, acc[s], false, false);
                }
            }
        }
#pragma unroll
        for (int s = 0; s < 8; ++s) {
            const int g = g0b + s * 16 + nB;
#pragma unroll
            for (int r = 0; r < 8; ++r)
                gatesLDS[(mBase + r) * G4 + g] = acc[s][r];
        }
        __syncthreads();

        // ================= layer 1 cell ==================================
#pragma unroll
        for (int b = 0; b < BT; ++b) {
            float pg[4];
#pragma unroll
            for (int g = 0; g < 4; ++g)
                pg[g] = gatesLDS[b * G4 + g * H + j] + b1c[g];
            float ig = 1.f / (1.f + __expf(-pg[0]));
            float fg = 1.f / (1.f + __expf(-pg[1]));
            float gg = tanhf(pg[2]);
            float og = 1.f / (1.f + __expf(-pg[3]));
            c1[b] = fg * c1[b] + ig * gg;
            float h = og * tanhf(c1[b]);
            h1t[b * H + j] = f2bf(h);
            if (t == TT - 1) hfin[b * H + j] = h;
        }
        __syncthreads();
    }

    // ---- output head: y = sigmoid(relu(h1_last) @ Wh^T + bh)
    if (tid < BT * 3) {
        int b = tid / 3, o = tid % 3;
        float s = bhb[o];
        for (int k = 0; k < H; ++k) {
            float hv = hfin[b * H + k];
            hv = hv > 0.f ? hv : 0.f;
            s += hv * Wh[o * H + k];
        }
        out[(bBase + b) * 3 + o] = 1.f / (1.f + __expf(-s));
    }
}

// ---------------------------------------------------------------------------
extern "C" void kernel_launch(void* const* d_in, const int* in_sizes, int n_in,
                              void* d_out, int out_size, void* d_ws, size_t ws_size,
                              hipStream_t stream) {
    const float* x    = (const float*)d_in[0];
    const float* Wih0 = (const float*)d_in[1];
    const float* Whh0 = (const float*)d_in[2];
    const float* bih0 = (const float*)d_in[3];
    const float* bhh0 = (const float*)d_in[4];
    const float* Wih1 = (const float*)d_in[5];
    const float* Whh1 = (const float*)d_in[6];
    const float* bih1 = (const float*)d_in[7];
    const float* bhh1 = (const float*)d_in[8];
    const float* Wh   = (const float*)d_in[9];
    const float* bh   = (const float*)d_in[10];

    u16* w16 = (u16*)d_ws;   // 3 * 1024*256 bf16 = 1.5 MB

    cvt_bf16_kernel<<<(3 * NWM + 255) / 256, 256, 0, stream>>>(Whh0, Wih1, Whh1, w16);

    lstm_persistent_kernel<<<256 / BT, 256, 0, stream>>>(
        x, Wih0, bih0, bhh0, bih1, bhh1, Wh, bh,
        w16, w16 + NWM, w16 + 2 * NWM, (float*)d_out);
}
